// FusedLlamaLowBitDecoderlayer_26895085207996
// MI455X (gfx1250) — matrix operand order repaired and checked by reading, not compiled, not run on hardware
//
#include <hip/hip_runtime.h>
#include <hip/hip_bf16.h>

// ---------------------------------------------------------------------------
// Fused Llama decoder (2 layers) for gfx1250 (MI455X, wave32):
//   * all GEMMs + attention through v_wmma_f32_16x16x32_bf16
//   * fp32 weights double-buffered through LDS with GLOBAL_LOAD_ASYNC_TO_LDS
//     (ASYNCcnt pipelining), shared by 4 waves -> 4x less HBM weight traffic
//   * flash attention with online softmax; V tiles async-staged to LDS
// ---------------------------------------------------------------------------

typedef __bf16 bf16_t;
typedef __attribute__((ext_vector_type(16))) __bf16 v16bf;
typedef __attribute__((ext_vector_type(8)))  __bf16 v8bf;
typedef __attribute__((ext_vector_type(8)))  float  v8f;
typedef __attribute__((ext_vector_type(4)))  float  v4f;
typedef int v4i_t __attribute__((vector_size(16)));   // matches builtin's V4i

static constexpr int S    = 1024;
static constexpr int H    = 4096;
static constexpr int NH   = 32;
static constexpr int NKV  = 8;
static constexpr int HD   = 128;
static constexpr int LNUM = 2;
static constexpr int IS   = 11008;
static constexpr float ATT_SCALE = 0.08838834764831845f; // 1/sqrt(128)

#define AS1 __attribute__((address_space(1)))
#define AS3 __attribute__((address_space(3)))

// --- CDNA5 async global->LDS copy (ASYNCcnt tracked), guarded fallbacks ----
__device__ inline void async_copy_b128(const float* g, float* l) {
#if __has_builtin(__builtin_amdgcn_global_load_async_to_lds_b128)
  __builtin_amdgcn_global_load_async_to_lds_b128(
      (AS1 v4i_t*)g, (AS3 v4i_t*)l, 0, 0);
#else
  *(v4f*)l = *(const v4f*)g;
#endif
}

template <int N>
__device__ inline void wait_asynccnt() {
#if __has_builtin(__builtin_amdgcn_s_wait_asynccnt)
  __builtin_amdgcn_s_wait_asynccnt(N);
#endif
}

__device__ inline v8f zero8() {
  v8f z;
#pragma unroll
  for (int i = 0; i < 8; ++i) z[i] = 0.0f;
  return z;
}

// 16 contiguous fp32 -> bf16 fragment (B-matrix lane data: K=klo..klo+15)
__device__ inline v16bf load16_f32_to_bf16(const float* p) {
  const v4f* p4 = (const v4f*)p;
  v4f a = p4[0], b = p4[1], c = p4[2], d = p4[3];
  v16bf r;
#pragma unroll
  for (int j = 0; j < 4; ++j) {
    r[j]      = (bf16_t)a[j];
    r[4 + j]  = (bf16_t)b[j];
    r[8 + j]  = (bf16_t)c[j];
    r[12 + j] = (bf16_t)d[j];
  }
  return r;
}

// A-matrix fragment from fp32 row: lane holds K = {0..15} rel. to p
__device__ inline v16bf load_a_frag_f32(const float* p) {
  const v4f* p4 = (const v4f*)p;
  v4f a = p4[0], b = p4[1], c = p4[2], d = p4[3];
  v16bf r;
#pragma unroll
  for (int j = 0; j < 4; ++j) {
    r[j]      = (bf16_t)a[j];
    r[4 + j]  = (bf16_t)b[j];
    r[8 + j]  = (bf16_t)c[j];
    r[12 + j] = (bf16_t)d[j];
  }
  return r;
}

// A-matrix fragment from bf16 row (two 16B loads)
__device__ inline v16bf load_a_frag_bf16(const bf16_t* p) {
  v8bf lo = *(const v8bf*)p;
  v8bf hi = *(const v8bf*)(p + 8);
  return __builtin_shufflevector(lo, hi, 0, 1, 2, 3, 4, 5, 6, 7,
                                 8, 9, 10, 11, 12, 13, 14, 15);
}

// ---------------------------------------------------------------------------
// RMSNorm: one block per row, outputs bf16 activations for WMMA A-operands.
// ---------------------------------------------------------------------------
__global__ __launch_bounds__(256)
void rmsnorm_kernel(const float* __restrict__ h, const float* __restrict__ w,
                    bf16_t* __restrict__ out) {
  __shared__ float red[256];
  const int row = blockIdx.x;
  const float* hr = h + (size_t)row * H;
  float s = 0.0f;
  for (int i = threadIdx.x; i < H; i += 256) {
    float v = hr[i];
    s += v * v;
  }
  red[threadIdx.x] = s;
  __syncthreads();
  for (int off = 128; off > 0; off >>= 1) {
    if (threadIdx.x < off) red[threadIdx.x] += red[threadIdx.x + off];
    __syncthreads();
  }
  const float rinv = rsqrtf(red[0] / (float)H + 1e-5f);
  bf16_t* orow = out + (size_t)row * H;
  for (int i = threadIdx.x; i < H; i += 256)
    orow[i] = (bf16_t)(w[i] * hr[i] * rinv);
}

// ---------------------------------------------------------------------------
// Weight tile stage: 64 rows x 32 k-floats -> LDS[64][36] (padded rows),
// 512 b128 async copies spread over 128 threads (4 per thread -> 4/wave).
// ---------------------------------------------------------------------------
__device__ inline void stage_wtile(const float* __restrict__ W, int ldw,
                                   int n0, int kk, float (*lds)[36], int tid) {
#pragma unroll
  for (int i = 0; i < 4; ++i) {
    const int idx = tid + i * 128;     // 0..511
    const int row = idx >> 3;          // 0..63
    const int c4  = (idx & 7) * 4;     // 0,4,...,28
    async_copy_b128(W + (size_t)(n0 + row) * ldw + kk + c4, &lds[row][c4]);
  }
}

// ---------------------------------------------------------------------------
// GEMM  C[M,N] (+= resid) = A[M,K](bf16) @ W[N,K]^T(fp32->bf16)
// Block = 64x64 output tile, 4 waves (one 16-row m-tile each) SHARING the
// weight tile, double-buffered in LDS via async copies.
// grid = (N/64, M/64)
// ---------------------------------------------------------------------------
template <bool RESID>
__global__ __launch_bounds__(128)
void gemm_bf16_kernel(const bf16_t* __restrict__ A, const float* __restrict__ W,
                      float* __restrict__ C, int M, int N, int K) {
  __shared__ float Wlds[2][64][36];

  const int tid  = threadIdx.x;
  const int lane = tid & 31;
  const int wave = tid >> 5;
  const int col  = lane & 15;
  const int half = lane >> 4;
  const int tm   = blockIdx.y * 4 + wave;
  const int n0   = blockIdx.x * 64;

  const bf16_t* Arow = A + (size_t)(tm * 16 + col) * K + half * 16;
  const int kb_off = half * 16;

  v8f acc[4];
#pragma unroll
  for (int sIdx = 0; sIdx < 4; ++sIdx) acc[sIdx] = zero8();

  stage_wtile(W, K, n0, 0, Wlds[0], tid);
  int ib = 0;
  for (int kk = 0; kk < K; kk += 32) {
    if (kk + 32 < K) {
      stage_wtile(W, K, n0, kk + 32, Wlds[ib ^ 1], tid);
      wait_asynccnt<4>();   // previous tile done, next tile in flight
    } else {
      wait_asynccnt<0>();
    }
    __syncthreads();

    v16bf a = load_a_frag_bf16(Arow + kk);
#pragma unroll
    for (int sIdx = 0; sIdx < 4; ++sIdx) {
      v16bf b = load16_f32_to_bf16(&Wlds[ib][sIdx * 16 + col][kb_off]);
      acc[sIdx] = __builtin_amdgcn_wmma_f32_16x16x32_bf16(
          false, a, false, b, (short)0, acc[sIdx], false, false);
    }
    __syncthreads();
    ib ^= 1;
  }

#pragma unroll
  for (int sIdx = 0; sIdx < 4; ++sIdx) {
    const int n = n0 + sIdx * 16 + col;
#pragma unroll
    for (int r = 0; r < 8; ++r) {
      const size_t idx = (size_t)(tm * 16 + half * 8 + r) * N + n;
      float v = acc[sIdx][r];
      if (RESID) v += C[idx];
      C[idx] = v;
    }
  }
}

// ---------------------------------------------------------------------------
// Fused gate/up GEMM + SiLU*up epilogue -> bf16.  Both weight tiles staged
// through double-buffered LDS; A-fragment reused across 8 WMMAs per k-step.
// ---------------------------------------------------------------------------
__global__ __launch_bounds__(128)
void gemm_gateup_kernel(const bf16_t* __restrict__ A,
                        const float* __restrict__ Wg,
                        const float* __restrict__ Wu,
                        bf16_t* __restrict__ out, int M, int N, int K) {
  __shared__ float Glds[2][64][36];
  __shared__ float Ulds[2][64][36];

  const int tid  = threadIdx.x;
  const int lane = tid & 31;
  const int wave = tid >> 5;
  const int col  = lane & 15;
  const int half = lane >> 4;
  const int tm   = blockIdx.y * 4 + wave;
  const int n0   = blockIdx.x * 64;

  const bf16_t* Arow = A + (size_t)(tm * 16 + col) * K + half * 16;
  const int kb_off = half * 16;

  v8f accg[4], accu[4];
#pragma unroll
  for (int sIdx = 0; sIdx < 4; ++sIdx) { accg[sIdx] = zero8(); accu[sIdx] = zero8(); }

  stage_wtile(Wg, K, n0, 0, Glds[0], tid);
  stage_wtile(Wu, K, n0, 0, Ulds[0], tid);
  int ib = 0;
  for (int kk = 0; kk < K; kk += 32) {
    if (kk + 32 < K) {
      stage_wtile(Wg, K, n0, kk + 32, Glds[ib ^ 1], tid);
      stage_wtile(Wu, K, n0, kk + 32, Ulds[ib ^ 1], tid);
      wait_asynccnt<8>();
    } else {
      wait_asynccnt<0>();
    }
    __syncthreads();

    v16bf a = load_a_frag_bf16(Arow + kk);
#pragma unroll
    for (int sIdx = 0; sIdx < 4; ++sIdx) {
      v16bf bg = load16_f32_to_bf16(&Glds[ib][sIdx * 16 + col][kb_off]);
      accg[sIdx] = __builtin_amdgcn_wmma_f32_16x16x32_bf16(
          false, a, false, bg, (short)0, accg[sIdx], false, false);
      v16bf bu = load16_f32_to_bf16(&Ulds[ib][sIdx * 16 + col][kb_off]);
      accu[sIdx] = __builtin_amdgcn_wmma_f32_16x16x32_bf16(
          false, a, false, bu, (short)0, accu[sIdx], false, false);
    }
    __syncthreads();
    ib ^= 1;
  }

#pragma unroll
  for (int sIdx = 0; sIdx < 4; ++sIdx) {
    const int n = n0 + sIdx * 16 + col;
#pragma unroll
    for (int r = 0; r < 8; ++r) {
      const size_t idx = (size_t)(tm * 16 + half * 8 + r) * N + n;
      float g = accg[sIdx][r];
      float u = accu[sIdx][r];
      float act = (g / (1.0f + __expf(-g))) * u;  // SiLU(g) * u
      out[idx] = (bf16_t)act;
    }
  }
}

// ---------------------------------------------------------------------------
// RoPE on Q (in place) and K (-> transposed KV cache in d_out), V transpose.
// ---------------------------------------------------------------------------
__global__ __launch_bounds__(256)
void rope_kernel(float* __restrict__ q, const float* __restrict__ kin,
                 const float* __restrict__ vin,
                 const float* __restrict__ cosT, const float* __restrict__ sinT,
                 const int* __restrict__ pos,
                 float* __restrict__ kout, float* __restrict__ vout) {
  const int s = blockIdx.x;
  const int p = pos[s];
  const float* cr = cosT + (size_t)p * HD;
  const float* sr = sinT + (size_t)p * HD;

  for (int i = threadIdx.x; i < NH * 64; i += blockDim.x) {
    const int hh = i >> 6;
    const int d  = i & 63;
    float* qp = q + (size_t)s * (NH * HD) + hh * HD;
    const float q1 = qp[d], q2 = qp[d + 64];
    qp[d]      = q1 * cr[d]      - q2 * sr[d];
    qp[d + 64] = q2 * cr[d + 64] + q1 * sr[d + 64];
  }
  for (int i = threadIdx.x; i < NKV * 64; i += blockDim.x) {
    const int hh = i >> 6;
    const int d  = i & 63;
    const float* kp = kin + (size_t)s * (NKV * HD) + hh * HD;
    const float k1 = kp[d], k2 = kp[d + 64];
    float* ko = kout + ((size_t)hh * S + s) * HD;
    ko[d]      = k1 * cr[d]      - k2 * sr[d];
    ko[d + 64] = k2 * cr[d + 64] + k1 * sr[d + 64];
  }
  for (int i = threadIdx.x; i < NKV * HD; i += blockDim.x) {
    const int hh = i >> 7;
    const int d  = i & 127;
    vout[((size_t)hh * S + s) * HD + d] = vin[(size_t)s * (NKV * HD) + hh * HD + d];
  }
}

// ---------------------------------------------------------------------------
// Flash attention (causal, GQA 4:1). Block = one head x 64 query rows,
// each wave a 16-row q-tile.  QK^T and P@V via WMMA; the shared 32x128 V tile
// is async-staged to LDS (overlapping the QK^T WMMAs); P converted
// C-layout -> A-layout through LDS.
// ---------------------------------------------------------------------------
__global__ __launch_bounds__(128)
void attn_kernel(const float* __restrict__ Q, const float* __restrict__ Kc,
                 const float* __restrict__ Vc, bf16_t* __restrict__ out) {
  __shared__ float  Vlds[32][HD];      // 16 KB, shared by the 4 waves
  __shared__ bf16_t Plds[4][16][32];   // 4 KB

  const int tid  = threadIdx.x;
  const int lane = tid & 31;
  const int wave = tid >> 5;
  const int col  = lane & 15;
  const int half = lane >> 4;
  const int head = blockIdx.x;
  const int kvh  = head >> 2;              // NH/NKV = 4
  const int q0   = blockIdx.y * 64 + wave * 16;

  const float* Krows = Kc + (size_t)kvh * S * HD;
  const float* Vrows = Vc + (size_t)kvh * S * HD;

  // Q fragments (16 rows x 128 dims -> 4 k-chunks of 32), loaded once
  const float* qp = Q + (size_t)(q0 + col) * (NH * HD) + head * HD;
  v16bf qf[4];
#pragma unroll
  for (int c = 0; c < 4; ++c) qf[c] = load_a_frag_f32(qp + c * 32 + half * 16);

  v8f oacc[8];
#pragma unroll
  for (int c = 0; c < 8; ++c) oacc[c] = zero8();
  float mrow[8], lrow[8];
#pragma unroll
  for (int r = 0; r < 8; ++r) { mrow[r] = -1e30f; lrow[r] = 0.0f; }

  const int kmax = blockIdx.y * 64 + 64;   // block-uniform causal bound

  for (int k0 = 0; k0 < kmax; k0 += 32) {
    // --- issue async V-tile stage (32 rows x 128 f32): overlaps QK^T ---
#pragma unroll
    for (int i = 0; i < 8; ++i) {
      const int idx = tid + i * 128;       // 0..1023
      const int row = idx >> 5;            // 0..31
      const int c4  = (idx & 31) * 4;      // 0..124
      async_copy_b128(Vrows + (size_t)(k0 + row) * HD + c4, &Vlds[row][c4]);
    }

    // --- scores for 32 keys (two 16x16 tiles) ---
    v8f sc0 = zero8(), sc1 = zero8();
#pragma unroll
    for (int c = 0; c < 4; ++c) {
      const int kb = c * 32 + half * 16;
      v16bf b0 = load16_f32_to_bf16(Krows + (size_t)(k0 + col) * HD + kb);
      sc0 = __builtin_amdgcn_wmma_f32_16x16x32_bf16(
          false, qf[c], false, b0, (short)0, sc0, false, false);
      v16bf b1 = load16_f32_to_bf16(Krows + (size_t)(k0 + 16 + col) * HD + kb);
      sc1 = __builtin_amdgcn_wmma_f32_16x16x32_bf16(
          false, qf[c], false, b1, (short)0, sc1, false, false);
    }

    // --- online softmax update (row stats reduced across 16-lane halves) ---
    float alpha[8];
#pragma unroll
    for (int r = 0; r < 8; ++r) {
      const int qi = q0 + half * 8 + r;
      float s0 = sc0[r] * ATT_SCALE;
      if (k0 + col > qi) s0 = -1e30f;
      float s1 = sc1[r] * ATT_SCALE;
      if (k0 + 16 + col > qi) s1 = -1e30f;
      float mx = fmaxf(s0, s1);
      mx = fmaxf(mx, __shfl_xor(mx, 1));
      mx = fmaxf(mx, __shfl_xor(mx, 2));
      mx = fmaxf(mx, __shfl_xor(mx, 4));
      mx = fmaxf(mx, __shfl_xor(mx, 8));
      const float mnew = fmaxf(mrow[r], mx);
      const float p0 = __expf(s0 - mnew);
      const float p1 = __expf(s1 - mnew);
      float ps = p0 + p1;
      ps += __shfl_xor(ps, 1);
      ps += __shfl_xor(ps, 2);
      ps += __shfl_xor(ps, 4);
      ps += __shfl_xor(ps, 8);
      alpha[r] = __expf(mrow[r] - mnew);
      lrow[r]  = lrow[r] * alpha[r] + ps;
      mrow[r]  = mnew;
      sc0[r] = p0;
      sc1[r] = p1;
    }
#pragma unroll
    for (int c = 0; c < 8; ++c)
#pragma unroll
      for (int r = 0; r < 8; ++r) oacc[c][r] *= alpha[r];

    // --- stage P (16x32 bf16) to LDS: C-layout -> row-major ---
#pragma unroll
    for (int r = 0; r < 8; ++r) {
      Plds[wave][half * 8 + r][col]      = (bf16_t)sc0[r];
      Plds[wave][half * 8 + r][16 + col] = (bf16_t)sc1[r];
    }
    wait_asynccnt<0>();   // V tile landed
    __syncthreads();      // P + V visible block-wide

    // --- reload P as A-fragment ---
    const bf16_t* pr = &Plds[wave][col][0];
    v8bf plo = *(const v8bf*)(pr + half * 16);
    v8bf phi = *(const v8bf*)(pr + half * 16 + 8);
    v16bf pa = __builtin_shufflevector(plo, phi, 0, 1, 2, 3, 4, 5, 6, 7,
                                       8, 9, 10, 11, 12, 13, 14, 15);

    // --- O += P @ V from LDS (8 dim-chunks of 16) ---
#pragma unroll
    for (int c = 0; c < 8; ++c) {
      v16bf vb;
#pragma unroll
      for (int j = 0; j < 16; ++j)
        vb[j] = (bf16_t)Vlds[half * 16 + j][c * 16 + col];
      oacc[c] = __builtin_amdgcn_wmma_f32_16x16x32_bf16(
          false, pa, false, vb, (short)0, oacc[c], false, false);
    }
    __syncthreads();
  }

  // --- epilogue: normalize and store bf16 for the O-projection GEMM ---
#pragma unroll
  for (int c = 0; c < 8; ++c) {
#pragma unroll
    for (int r = 0; r < 8; ++r) {
      const int qi = q0 + half * 8 + r;
      const float o = oacc[c][r] / lrow[r];
      out[(size_t)qi * (NH * HD) + head * HD + c * 16 + col] = (bf16_t)o;
    }
  }
}

// ---------------------------------------------------------------------------
// Host orchestration
// ---------------------------------------------------------------------------
extern "C" void kernel_launch(void* const* d_in, const int* in_sizes, int n_in,
                              void* d_out, int out_size, void* d_ws, size_t ws_size,
                              hipStream_t stream) {
  (void)in_sizes; (void)n_in; (void)out_size; (void)ws_size;

  const float* x    = (const float*)d_in[0];
  const int*   pos  = (const int*)d_in[2];
  const float* ccos = (const float*)d_in[3];
  const float* csin = (const float*)d_in[4];
  const float* ln1  = (const float*)d_in[5];
  const float* ln2  = (const float*)d_in[6];
  const float* qw   = (const float*)d_in[7];
  const float* kw   = (const float*)d_in[8];
  const float* vw   = (const float*)d_in[9];
  const float* ow   = (const float*)d_in[10];
  const float* gw   = (const float*)d_in[11];
  const float* uw   = (const float*)d_in[12];
  const float* dw   = (const float*)d_in[13];

  float* h    = (float*)d_out;                       // [S][H] running hidden
  float* keys = h + (size_t)S * H;                   // [L][NKV][S][HD]
  float* vals = keys + (size_t)LNUM * NKV * S * HD;  // [L][NKV][S][HD]

  char*   wsb     = (char*)d_ws;
  size_t  off     = 0;
  bf16_t* ws_n    = (bf16_t*)(wsb + off); off += (size_t)S * H * sizeof(bf16_t);
  float*  ws_q    = (float*)(wsb + off);  off += (size_t)S * H * sizeof(float);
  float*  ws_k    = (float*)(wsb + off);  off += (size_t)S * NKV * HD * sizeof(float);
  float*  ws_v    = (float*)(wsb + off);  off += (size_t)S * NKV * HD * sizeof(float);
  bf16_t* ws_attn = (bf16_t*)(wsb + off); off += (size_t)S * H * sizeof(bf16_t);
  bf16_t* ws_act  = (bf16_t*)(wsb + off);

  (void)hipMemcpyAsync(h, x, (size_t)S * H * sizeof(float),
                       hipMemcpyDeviceToDevice, stream);

  for (int l = 0; l < LNUM; ++l) {
    const float* qwl = qw + (size_t)l * NH * HD * H;
    const float* kwl = kw + (size_t)l * NKV * HD * H;
    const float* vwl = vw + (size_t)l * NKV * HD * H;
    const float* owl = ow + (size_t)l * H * (NH * HD);
    const float* gwl = gw + (size_t)l * IS * H;
    const float* uwl = uw + (size_t)l * IS * H;
    const float* dwl = dw + (size_t)l * H * IS;
    float* kol = keys + (size_t)l * NKV * S * HD;
    float* vol = vals + (size_t)l * NKV * S * HD;

    // attention block
    rmsnorm_kernel<<<S, 256, 0, stream>>>(h, ln1 + (size_t)l * H, ws_n);
    gemm_bf16_kernel<false><<<dim3(H / 64, S / 64), 128, 0, stream>>>(
        ws_n, qwl, ws_q, S, H, H);
    gemm_bf16_kernel<false><<<dim3((NKV * HD) / 64, S / 64), 128, 0, stream>>>(
        ws_n, kwl, ws_k, S, NKV * HD, H);
    gemm_bf16_kernel<false><<<dim3((NKV * HD) / 64, S / 64), 128, 0, stream>>>(
        ws_n, vwl, ws_v, S, NKV * HD, H);
    rope_kernel<<<S, 256, 0, stream>>>(ws_q, ws_k, ws_v, ccos, csin, pos,
                                       kol, vol);
    attn_kernel<<<dim3(NH, S / 64), 128, 0, stream>>>(ws_q, kol, vol, ws_attn);
    gemm_bf16_kernel<true><<<dim3(H / 64, S / 64), 128, 0, stream>>>(
        ws_attn, owl, h, S, H, H);

    // MLP block
    rmsnorm_kernel<<<S, 256, 0, stream>>>(h, ln2 + (size_t)l * H, ws_n);
    gemm_gateup_kernel<<<dim3(IS / 64, S / 64), 128, 0, stream>>>(
        ws_n, gwl, uwl, ws_act, S, IS, H);
    gemm_bf16_kernel<true><<<dim3(H / 64, S / 64), 128, 0, stream>>>(
        ws_act, dwl, h, S, H, IS);
  }
}
